// LinearRecurrentUnitFFN_12799002542675
// MI455X (gfx1250) — compile-verified, matrix operand order checked by reading, not run on hardware
//
#include <hip/hip_runtime.h>
#include <hip/hip_bf16.h>

// ---------------------------------------------------------------------------
// LRU-FFN:  y = Re( scan(λ, u @ (γB)) @ C ) + D·u
//   GEMM1/GEMM2: v_wmma_f32_16x16x32_bf16, 4 M-tiles per wave,
//     block-shared weight panel staged in LDS (ds_load_b128, padded rows)
//   gamma folded into bf16(B); u pre-converted to bf16 (L2-resident re-reads)
//   scan: 3-phase chunked recurrence, f32 accumulate, bf16 storage, in-place
// ---------------------------------------------------------------------------

typedef __attribute__((ext_vector_type(16))) __bf16       v16bf;
typedef __attribute__((ext_vector_type(8)))  float        v8f;
typedef __attribute__((ext_vector_type(4)))  unsigned int u32x4;
typedef __attribute__((ext_vector_type(4)))  float        f32x4;
typedef __attribute__((ext_vector_type(4)))  __bf16       v4bf;

#define BATCH 8
#define TLEN  2048
#define NDIM  512
#define MROWS (BATCH * TLEN)      // 16384
#define NCH   32                  // scan chunks
#define CHL   (TLEN / NCH)        // 64 steps per chunk
#define SROW  520                 // LDS row pitch (512 + 8 pad -> 4-dword bank skew)

union Frag16 { v16bf v; u32x4 u[2]; };

__device__ __forceinline__ __bf16 tobf(float f) { return (__bf16)f; }

// --------------------------- setup: λ and λ^CHL -----------------------------
__global__ void lru_setup(const float* __restrict__ nu, const float* __restrict__ theta,
                          float* lamr, float* lami, float* lamLr, float* lamLi) {
  int n = threadIdx.x;
  float e  = __expf(nu[n]);
  float r  = __expf(-e);
  float th = theta[n];
  lamr[n] = r * __cosf(th);
  lami[n] = r * __sinf(th);
  float rl  = __expf(-(float)CHL * e);
  float thl = (float)CHL * th;
  lamLr[n] = rl * __cosf(thl);
  lamLi[n] = rl * __sinf(thl);
}

// ---- convert + transpose to bf16 [n][k];  gamma folded into B here ---------
__global__ void lru_convertBC(const float* __restrict__ Bre, const float* __restrict__ Bim,
                              const float* __restrict__ Cre, const float* __restrict__ Cim,
                              const float* __restrict__ gamma,
                              __bf16* BreT, __bf16* BimT, __bf16* CreT, __bf16* CimT) {
  int idx = blockIdx.x * blockDim.x + threadIdx.x;   // idx = k*512+n
  int k = idx >> 9;
  int n = idx & 511;
  int s = n * NDIM + k;                              // transposed slot
  float g = gamma[k];
  BreT[s] = tobf(g * Bre[idx]);
  BimT[s] = tobf(g * Bim[idx]);
  CreT[s] = tobf(Cre[idx]);
  CimT[s] = tobf(Cim[idx]);
}

// ----------------------- convert u -> bf16 (streaming) ----------------------
__global__ __launch_bounds__(256)
void lru_convert_u(const float* __restrict__ u, __bf16* __restrict__ ubf) {
  size_t i4 = ((size_t)blockIdx.x * blockDim.x + threadIdx.x) * 4;
  f32x4 f = *(const f32x4*)(u + i4);
  v4bf o;
  o[0] = tobf(f[0]); o[1] = tobf(f[1]); o[2] = tobf(f[2]); o[3] = tobf(f[3]);
  *(v4bf*)(ubf + i4) = o;
}

// ---- cooperative: stage a 16-col x 512-K bf16 weight panel pair into LDS ---
__device__ __forceinline__ void stage_panel(const __bf16* __restrict__ g0,
                                            const __bf16* __restrict__ g1,
                                            __bf16* s0, __bf16* s1, int col0) {
  for (int c = threadIdx.x; c < 16 * 64; c += 256) {   // 1024 x 16B chunks / matrix
    int row = c >> 6;                                  // 0..15
    int pos = (c & 63) * 8;                            // 0..504
    u32x4 vr = *(const u32x4*)&g0[(size_t)(col0 + row) * NDIM + pos];
    u32x4 vi = *(const u32x4*)&g1[(size_t)(col0 + row) * NDIM + pos];
    *(u32x4*)&s0[row * SROW + pos] = vr;
    *(u32x4*)&s1[row * SROW + pos] = vi;
  }
  __syncthreads();
}

// ------ GEMM1: new_u_{re,im} = ubf @ (γB)_{re,im}; 64x16 strip per wave -----
__global__ __launch_bounds__(256)
void lru_gemm1(const __bf16* __restrict__ ubf,
               const __bf16* __restrict__ BreT, const __bf16* __restrict__ BimT,
               __bf16* __restrict__ nubr, __bf16* __restrict__ nubi) {
  __shared__ __bf16 sBr[16 * SROW];
  __shared__ __bf16 sBi[16 * SROW];
  const int lane  = threadIdx.x & 31;
  const int wave  = threadIdx.x >> 5;
  const int col0  = blockIdx.x * 16;
  const int row0  = (blockIdx.y * 8 + wave) * 64;     // 4 x 16-row tiles
  const int rlo   = lane & 15;
  const int khalf = (lane >> 4) * 8;                  // ISA 16-bit K-half select

  stage_panel(BreT, BimT, sBr, sBi, col0);

  const __bf16* sbr = &sBr[rlo * SROW + khalf];
  const __bf16* sbi = &sBi[rlo * SROW + khalf];
  const __bf16* pa  = &ubf[((size_t)row0 + rlo) * NDIM + khalf];

  v8f accr[4] = {{}, {}, {}, {}};
  v8f acci[4] = {{}, {}, {}, {}};
  for (int k0 = 0; k0 < NDIM; k0 += 32) {
    Frag16 br, bi;
    br.u[0] = *(const u32x4*)(sbr + k0);  br.u[1] = *(const u32x4*)(sbr + k0 + 16);
    bi.u[0] = *(const u32x4*)(sbi + k0);  bi.u[1] = *(const u32x4*)(sbi + k0 + 16);
#pragma unroll
    for (int mt = 0; mt < 4; ++mt) {
      const __bf16* p = pa + (size_t)mt * 16 * NDIM + k0;
      __builtin_prefetch((const void*)(p + 32), 0, 1);   // next k-step A
      Frag16 a;
      a.u[0] = *(const u32x4*)p;  a.u[1] = *(const u32x4*)(p + 16);
      accr[mt] = __builtin_amdgcn_wmma_f32_16x16x32_bf16(false, a.v, false, br.v,
                                                         (short)0, accr[mt], false, false);
      acci[mt] = __builtin_amdgcn_wmma_f32_16x16x32_bf16(false, a.v, false, bi.v,
                                                         (short)0, acci[mt], false, false);
    }
  }

  const int ocol = col0 + rlo;
  const int rup  = (lane >> 4) * 8;                   // C/D layout row offset
#pragma unroll
  for (int mt = 0; mt < 4; ++mt) {
#pragma unroll
    for (int r = 0; r < 8; ++r) {
      size_t oi = (size_t)(row0 + mt * 16 + rup + r) * NDIM + ocol;
      nubr[oi] = tobf(accr[mt][r]);
      nubi[oi] = tobf(acci[mt][r]);
    }
  }
}

// ----------------- scan phase 1: local chunk scan, in place -----------------
__global__ __launch_bounds__(512)
void lru_scan_local(__bf16* __restrict__ nubr, __bf16* __restrict__ nubi,
                    const float* __restrict__ lamr, const float* __restrict__ lami,
                    float* __restrict__ carr, float* __restrict__ cari) {
  const int b = blockIdx.x / NCH;
  const int c = blockIdx.x % NCH;
  const int n = threadIdx.x;
  const float lr = lamr[n], li = lami[n];
  float xr = 0.f, xi = 0.f;
  size_t idx = ((size_t)b * TLEN + (size_t)c * CHL) * NDIM + n;
  for (int j = 0; j < CHL; ++j, idx += NDIM) {
    float ar = (float)nubr[idx], ai = (float)nubi[idx];
    float nxr = lr * xr - li * xi + ar;
    float nxi = lr * xi + li * xr + ai;
    xr = nxr; xi = nxi;
    nubr[idx] = tobf(xr); nubi[idx] = tobf(xi);
  }
  carr[((size_t)b * NCH + c) * NDIM + n] = xr;
  cari[((size_t)b * NCH + c) * NDIM + n] = xi;
}

// --------- scan phase 2: serial combine of chunk carries with λ^CHL ---------
__global__ __launch_bounds__(512)
void lru_scan_carry(float* __restrict__ carr, float* __restrict__ cari,
                    const float* __restrict__ lamLr, const float* __restrict__ lamLi) {
  const int b = blockIdx.x;
  const int n = threadIdx.x;
  const float lr = lamLr[n], li = lamLi[n];
  float xr = 0.f, xi = 0.f;
  for (int c = 0; c < NCH; ++c) {
    size_t idx = ((size_t)b * NCH + c) * NDIM + n;
    float cr = carr[idx], ci = cari[idx];
    float nxr = lr * xr - li * xi + cr;
    float nxi = lr * xi + li * xr + ci;
    xr = nxr; xi = nxi;
    carr[idx] = xr; cari[idx] = xi;      // true state after chunk c
  }
}

// - scan phase 3: x[t] = local[t] + λ^{j+1}·X_{c-1}; in-place emit (x_re,-x_im)
__global__ __launch_bounds__(512)
void lru_scan_fixup(__bf16* __restrict__ nubr, __bf16* __restrict__ nubi,
                    const float* __restrict__ carr, const float* __restrict__ cari,
                    const float* __restrict__ lamr, const float* __restrict__ lami) {
  const int b = blockIdx.x / NCH;
  const int c = blockIdx.x % NCH;
  const int n = threadIdx.x;
  float Xr = 0.f, Xi = 0.f;
  if (c > 0) {
    size_t ci = ((size_t)b * NCH + (c - 1)) * NDIM + n;
    Xr = carr[ci]; Xi = cari[ci];
  }
  const float lr = lamr[n], li = lami[n];
  float pr = lr, pi = li;                              // λ^{j+1}
  size_t idx = ((size_t)b * TLEN + (size_t)c * CHL) * NDIM + n;
  for (int j = 0; j < CHL; ++j, idx += NDIM) {
    float xr = (float)nubr[idx] + (pr * Xr - pi * Xi);
    float xi = (float)nubi[idx] + (pr * Xi + pi * Xr);
    nubr[idx] = tobf(xr);
    nubi[idx] = tobf(-xi);                             // pre-negated for GEMM2
    float npr = pr * lr - pi * li;
    float npi = pr * li + pi * lr;
    pr = npr; pi = npi;
  }
}

// -- GEMM2: y = x_re@C_re + (-x_im)@C_im + D·u;  64x16 strip per wave --------
__global__ __launch_bounds__(256)
void lru_gemm2(const __bf16* __restrict__ xre, const __bf16* __restrict__ ximn,
               const __bf16* __restrict__ CreT, const __bf16* __restrict__ CimT,
               const float* __restrict__ u, const float* __restrict__ D,
               float* __restrict__ y) {
  __shared__ __bf16 sCr[16 * SROW];
  __shared__ __bf16 sCi[16 * SROW];
  const int lane  = threadIdx.x & 31;
  const int wave  = threadIdx.x >> 5;
  const int col0  = blockIdx.x * 16;
  const int row0  = (blockIdx.y * 8 + wave) * 64;
  const int rlo   = lane & 15;
  const int khalf = (lane >> 4) * 8;

  stage_panel(CreT, CimT, sCr, sCi, col0);

  const __bf16* scr = &sCr[rlo * SROW + khalf];
  const __bf16* sci = &sCi[rlo * SROW + khalf];
  const __bf16* par = &xre [((size_t)row0 + rlo) * NDIM + khalf];
  const __bf16* pai = &ximn[((size_t)row0 + rlo) * NDIM + khalf];

  v8f acc[4] = {{}, {}, {}, {}};
  for (int k0 = 0; k0 < NDIM; k0 += 32) {
    Frag16 cr, ci;
    cr.u[0] = *(const u32x4*)(scr + k0);  cr.u[1] = *(const u32x4*)(scr + k0 + 16);
    ci.u[0] = *(const u32x4*)(sci + k0);  ci.u[1] = *(const u32x4*)(sci + k0 + 16);
#pragma unroll
    for (int mt = 0; mt < 4; ++mt) {
      const __bf16* pr0 = par + (size_t)mt * 16 * NDIM + k0;
      const __bf16* pi0 = pai + (size_t)mt * 16 * NDIM + k0;
      __builtin_prefetch((const void*)(pr0 + 32), 0, 1);
      __builtin_prefetch((const void*)(pi0 + 32), 0, 1);
      Frag16 ar, ai;
      ar.u[0] = *(const u32x4*)pr0;  ar.u[1] = *(const u32x4*)(pr0 + 16);
      ai.u[0] = *(const u32x4*)pi0;  ai.u[1] = *(const u32x4*)(pi0 + 16);
      acc[mt] = __builtin_amdgcn_wmma_f32_16x16x32_bf16(false, ar.v, false, cr.v,
                                                        (short)0, acc[mt], false, false);
      acc[mt] = __builtin_amdgcn_wmma_f32_16x16x32_bf16(false, ai.v, false, ci.v,
                                                        (short)0, acc[mt], false, false);
    }
  }

  const int ocol = col0 + rlo;
  const int rup  = (lane >> 4) * 8;
  const float d  = D[ocol];
#pragma unroll
  for (int mt = 0; mt < 4; ++mt) {
#pragma unroll
    for (int r = 0; r < 8; ++r) {
      size_t oi = (size_t)(row0 + mt * 16 + rup + r) * NDIM + ocol;
      y[oi] = acc[mt][r] + d * u[oi];
    }
  }
}

// ---------------------------------------------------------------------------
extern "C" void kernel_launch(void* const* d_in, const int* in_sizes, int n_in,
                              void* d_out, int out_size, void* d_ws, size_t ws_size,
                              hipStream_t stream) {
  const float* u     = (const float*)d_in[0];
  const float* Cre   = (const float*)d_in[1];
  const float* Cim   = (const float*)d_in[2];
  const float* Bre   = (const float*)d_in[3];
  const float* Bim   = (const float*)d_in[4];
  const float* D     = (const float*)d_in[5];
  const float* nu    = (const float*)d_in[6];
  const float* theta = (const float*)d_in[7];
  const float* gamma = (const float*)d_in[8];
  float* y = (float*)d_out;

  char* ws = (char*)d_ws;
  constexpr size_t MN     = (size_t)MROWS * NDIM;      // 8,388,608
  constexpr size_t MAT_BF = (size_t)NDIM * NDIM * 2;   // 512 KiB per bf16 matrix
  constexpr size_t CAR    = (size_t)BATCH * NCH * NDIM * 4;

  size_t off = 0;
  float* lamr  = (float*)(ws + off); off += NDIM * 4;
  float* lami  = (float*)(ws + off); off += NDIM * 4;
  float* lamLr = (float*)(ws + off); off += NDIM * 4;
  float* lamLi = (float*)(ws + off); off += NDIM * 4;
  off = 8192;                                          // align matrix region
  __bf16* BreT = (__bf16*)(ws + off); off += MAT_BF;
  __bf16* BimT = (__bf16*)(ws + off); off += MAT_BF;
  __bf16* CreT = (__bf16*)(ws + off); off += MAT_BF;
  __bf16* CimT = (__bf16*)(ws + off); off += MAT_BF;
  __bf16* ubf  = (__bf16*)(ws + off); off += MN * 2;
  __bf16* nubr = (__bf16*)(ws + off); off += MN * 2;   // new_u re -> local -> x_re
  __bf16* nubi = (__bf16*)(ws + off); off += MN * 2;   // new_u im -> local -> -x_im
  float*  carr = (float*)(ws + off);  off += CAR;
  float*  cari = (float*)(ws + off);  off += CAR;

  lru_setup<<<1, NDIM, 0, stream>>>(nu, theta, lamr, lami, lamLr, lamLi);
  lru_convertBC<<<(NDIM * NDIM) / 256, 256, 0, stream>>>(Bre, Bim, Cre, Cim, gamma,
                                                         BreT, BimT, CreT, CimT);
  lru_convert_u<<<(int)(MN / 4 / 256), 256, 0, stream>>>(u, ubf);
  dim3 ggrid(NDIM / 16, MROWS / 512);                  // (32, 32): 8 waves x 64 rows
  lru_gemm1<<<ggrid, 256, 0, stream>>>(ubf, BreT, BimT, nubr, nubi);
  lru_scan_local<<<BATCH * NCH, NDIM, 0, stream>>>(nubr, nubi, lamr, lami, carr, cari);
  lru_scan_carry<<<BATCH, NDIM, 0, stream>>>(carr, cari, lamLr, lamLi);
  lru_scan_fixup<<<BATCH * NCH, NDIM, 0, stream>>>(nubr, nubi, carr, cari, lamr, lami);
  lru_gemm2<<<ggrid, 256, 0, stream>>>(nubr, nubi, CreT, CimT, u, D, y);
}
